// MultilayerGRU_16243566314000
// MI455X (gfx1250) — compile-verified, matrix operand order checked by reading, not run on hardware
//
#include <hip/hip_runtime.h>

// ---------------------------------------------------------------------------
// Multilayer GRU (L=2, B=32, S=512, I=256, H=1024, O=256) for gfx1250.
// bf16 WMMA (v_wmma_f32_16x16x32_bf16), persistent kernel, grid barriers.
// ---------------------------------------------------------------------------

typedef __attribute__((ext_vector_type(16))) __bf16 v16bf;
typedef __attribute__((ext_vector_type(8)))  __bf16 v8bf;
typedef __attribute__((ext_vector_type(8)))  float  v8f;

#define B_ 32
#define S_ 512
#define I_ 256
#define H_ 1024
#define O_ 256
#define L_ 2

#define NBLOCKS  32
#define NTHREADS 256

// ------------------------- workspace layout (bytes) ------------------------
// all sizes are multiples of 256 so every region stays 256B-aligned
static const size_t OFF_BAR = 0;                                   // 2 u32
static const size_t OFF_WX0 = 256;                                 // 3*H*I bf16
static const size_t OFF_WX1 = OFF_WX0 + (size_t)3*H_*I_*2;         // 3*H*H bf16
static const size_t OFF_WH  = OFF_WX1 + (size_t)3*H_*H_*2;         // 2*3*H*H bf16
static const size_t OFF_WHY = OFF_WH  + (size_t)6*H_*H_*2;         // O*H bf16
static const size_t OFF_XP  = OFF_WHY + (size_t)O_*H_*2;           // B*S*I bf16
static const size_t OFF_HF  = OFF_XP  + (size_t)B_*S_*I_*2;        // 2*B*H f32
static const size_t OFF_HB  = OFF_HF  + (size_t)L_*B_*H_*4;        // 2*B*H bf16
static const size_t OFF_Z   = OFF_HB  + (size_t)L_*B_*H_*2;        // B*H f32
static const size_t OFF_RH  = OFF_Z   + (size_t)B_*H_*4;           // B*H bf16

// ------------------------------- helpers -----------------------------------
static __device__ __forceinline__ int laneId() { return (int)(threadIdx.x & 31u); }

static __device__ __forceinline__ v8f wmma_bf16(v16bf a, v16bf b, v8f c) {
  // D(f32 16x16) = A(bf16 16x32) * B(bf16 32x16) + C
  return __builtin_amdgcn_wmma_f32_16x16x32_bf16(false, a, false, b, (short)0, c,
                                                 false, false);
}

// Fragment-packed operand: 512 bf16 per fragment, lane-major (lane*16 .. +15).
// One 32-byte contiguous load per lane -> two global_load_b128.
static __device__ __forceinline__ v16bf load_pack(const __bf16* __restrict__ pk,
                                                  int frag) {
  const __bf16* p = pk + ((size_t)frag << 9) + (laneId() << 4);
  return *(const v16bf*)p;
}

// A fragment (16x32 bf16) gathered from a row-major [*, ld] bf16 matrix.
// Per ISA layout: lane l -> row m0+(l&15); k-base = k0 + 8*(l>=16);
// VGPRs 0..3 hold k..k+7, VGPRs 4..7 hold k+16..k+23  -> two aligned b128 loads.
static __device__ __forceinline__ v16bf load_afrag(const __bf16* __restrict__ base,
                                                   int ld, int m0, int k0) {
  int l  = laneId();
  int m  = m0 + (l & 15);
  int kb = k0 + ((l >> 4) << 3);
  const __bf16* p = base + (size_t)m * ld + kb;
  v8bf lo = *(const v8bf*)p;          // k .. k+7
  v8bf hi = *(const v8bf*)(p + 16);   // k+16 .. k+23
  return __builtin_shufflevector(lo, hi, 0,1,2,3,4,5,6,7,8,9,10,11,12,13,14,15);
}

static __device__ __forceinline__ float fsigmoid(float x) {
  return 1.0f / (1.0f + __expf(-x));
}
static __device__ __forceinline__ float ftanh(float x) {
  return 1.0f - 2.0f / (1.0f + __expf(2.0f * x));
}

// Sense-reversing grid barrier (32 blocks, guaranteed co-resident).
static __device__ __forceinline__ void grid_sync(volatile unsigned* arrive,
                                                 volatile unsigned* gen) {
  __syncthreads();
  if (threadIdx.x == 0) {
    __threadfence();
    unsigned g = *gen;
    if (atomicAdd((unsigned*)arrive, 1u) == (unsigned)(NBLOCKS - 1)) {
      *arrive = 0u;
      __threadfence();
      *gen = g + 1u;
    } else {
      while (*gen == g) { __builtin_amdgcn_s_sleep(2); }
    }
    __threadfence();
  }
  __syncthreads();
}

// ----------------------------- setup kernels -------------------------------
// Pack a row-major fp32 weight [N_total, K] into bf16 B-operand fragments.
__global__ void pack_b_kernel(const float* __restrict__ w, __bf16* __restrict__ dst,
                              int K, int KT, long total) {
  long stride = (long)gridDim.x * blockDim.x;
  for (long p = (long)blockIdx.x * blockDim.x + threadIdx.x; p < total; p += stride) {
    int  e2   = (int)(p & 15);
    int  l    = (int)((p >> 4) & 31);
    long frag = p >> 9;
    int  kt   = (int)(frag % KT);
    long nt   = frag / KT;
    int  v = e2 >> 1, e = e2 & 1;
    long n = nt * 16 + (l & 15);
    int  k = kt * 32 + ((l >> 4) << 3) + ((v >= 4) ? 16 : 0) + ((v & 3) << 1) + e;
    dst[p] = (__bf16)w[n * K + k];
  }
}

// Pack x [B,S,I] fp32 into per-(s,mt,kt) bf16 A-operand fragments.
__global__ void pack_x_kernel(const float* __restrict__ x, __bf16* __restrict__ xp,
                              long total) {
  long stride = (long)gridDim.x * blockDim.x;
  for (long p = (long)blockIdx.x * blockDim.x + threadIdx.x; p < total; p += stride) {
    int  e2   = (int)(p & 15);
    int  l    = (int)((p >> 4) & 31);
    long frag = p >> 9;
    int  kt   = (int)(frag & 7);
    long rest = frag >> 3;
    int  mt   = (int)(rest & 1);
    int  s    = (int)(rest >> 1);
    int  v = e2 >> 1, e = e2 & 1;
    int  m = mt * 16 + (l & 15);
    int  k = kt * 32 + ((l >> 4) << 3) + ((v >= 4) ? 16 : 0) + ((v & 3) << 1) + e;
    xp[p] = (__bf16)x[(size_t)m * (S_ * I_) + (size_t)s * I_ + k];
  }
}

// Reset barrier + initialize hidden state (fp32 master + bf16 mirror).
__global__ void init_kernel(const float* __restrict__ h0, float* __restrict__ hf,
                            __bf16* __restrict__ hb, unsigned* __restrict__ bar) {
  if (blockIdx.x == 0 && threadIdx.x < 2) bar[threadIdx.x] = 0u;
  int stride = gridDim.x * blockDim.x;
  for (int i = blockIdx.x * blockDim.x + threadIdx.x; i < L_ * B_ * H_; i += stride) {
    int ly = i >> 15;            // hf layout [layer][b][h], 32*1024 = 2^15
    int b  = (i >> 10) & 31;
    int h  = i & (H_ - 1);
    float v = h0[(size_t)b * (L_ * H_) + (size_t)ly * H_ + h];
    hf[i] = v;
    hb[i] = (__bf16)v;
  }
}

// --------------------------- persistent GRU kernel -------------------------
__global__ void __launch_bounds__(NTHREADS)
gru_persistent(const __bf16* __restrict__ wx0p, const __bf16* __restrict__ wx1p,
               const __bf16* __restrict__ whp,  const __bf16* __restrict__ whyp,
               const __bf16* __restrict__ xp,
               float* __restrict__ hf, __bf16* __restrict__ hb,
               float* __restrict__ zbuf, __bf16* __restrict__ rh,
               unsigned* __restrict__ bar,
               const float* __restrict__ bh, const float* __restrict__ by,
               float* __restrict__ out) {
  const int wid = (int)((blockIdx.x * blockDim.x + threadIdx.x) >> 5);
  const int nw  = (NBLOCKS * NTHREADS) >> 5;
  volatile unsigned* arrive = bar;
  volatile unsigned* gen    = bar + 1;

  const int l     = laneId();
  const int nloc  = l & 15;            // C/D fragment: n = lane&15
  const int mbase = (l >> 4) << 3;     // C/D fragment: m = 8*(lane>=16) + vgpr

  float* hf0 = hf;                 __bf16* hb0 = hb;
  float* hf1 = hf + B_ * H_;       __bf16* hb1 = hb + B_ * H_;
  const __bf16* wh0 = whp;                       // layer 0: [3H,H] fragments
  const __bf16* wh1 = whp + (size_t)3 * H_ * H_; // layer 1

  for (int s = 0; s < S_; ++s) {
    // ---- Phase A: layer0 z,r  (A=[x_t | h0], K=1280, N=2048) -------------
    for (int job = wid; job < 2 * 128; job += nw) {
      int mt = job & 1, nt = job >> 1;
      v8f acc = {};
      #pragma unroll
      for (int f = 0; f < 8; ++f)
        acc = wmma_bf16(load_pack(xp, (s * 2 + mt) * 8 + f),
                        load_pack(wx0p, nt * 8 + f), acc);
      #pragma unroll 4
      for (int kk = 0; kk < 32; ++kk)
        acc = wmma_bf16(load_afrag(hb0, H_, mt * 16, kk * 32),
                        load_pack(wh0, nt * 32 + kk), acc);
      int col3 = nt * 16 + nloc;
      int gate = col3 >> 10, hc = col3 & (H_ - 1);
      float bias = bh[gate * H_ + hc];
      #pragma unroll
      for (int j = 0; j < 8; ++j) {
        int b = mt * 16 + mbase + j;
        float sg = fsigmoid(acc[j] + bias);
        if (gate == 0) zbuf[b * H_ + hc] = sg;
        else           rh[b * H_ + hc] = (__bf16)(sg * hf0[b * H_ + hc]);
      }
    }
    grid_sync(arrive, gen);

    // ---- Phase B: layer0 g + state update  (A=[x_t | r*h0], N=1024) ------
    for (int job = wid; job < 2 * 64; job += nw) {
      int mt = job & 1, nt3 = 128 + (job >> 1);
      v8f acc = {};
      #pragma unroll
      for (int f = 0; f < 8; ++f)
        acc = wmma_bf16(load_pack(xp, (s * 2 + mt) * 8 + f),
                        load_pack(wx0p, nt3 * 8 + f), acc);
      #pragma unroll 4
      for (int kk = 0; kk < 32; ++kk)
        acc = wmma_bf16(load_afrag(rh, H_, mt * 16, kk * 32),
                        load_pack(wh0, nt3 * 32 + kk), acc);
      int hc = (nt3 - 128) * 16 + nloc;
      float bias = bh[2 * H_ + hc];
      #pragma unroll
      for (int j = 0; j < 8; ++j) {
        int b = mt * 16 + mbase + j;
        float g  = ftanh(acc[j] + bias);
        float z  = zbuf[b * H_ + hc];
        float hn = z * hf0[b * H_ + hc] + (1.0f - z) * g;
        hf0[b * H_ + hc] = hn;
        hb0[b * H_ + hc] = (__bf16)hn;
      }
    }
    grid_sync(arrive, gen);

    // ---- Phase C: layer1 z,r  (A=[h0_new | h1], K=2048, N=2048) ----------
    for (int job = wid; job < 2 * 128; job += nw) {
      int mt = job & 1, nt = job >> 1;
      v8f acc = {};
      #pragma unroll 4
      for (int f = 0; f < 32; ++f)
        acc = wmma_bf16(load_afrag(hb0, H_, mt * 16, f * 32),
                        load_pack(wx1p, nt * 32 + f), acc);
      #pragma unroll 4
      for (int kk = 0; kk < 32; ++kk)
        acc = wmma_bf16(load_afrag(hb1, H_, mt * 16, kk * 32),
                        load_pack(wh1, nt * 32 + kk), acc);
      int col3 = nt * 16 + nloc;
      int gate = col3 >> 10, hc = col3 & (H_ - 1);
      float bias = bh[(3 + gate) * H_ + hc];
      #pragma unroll
      for (int j = 0; j < 8; ++j) {
        int b = mt * 16 + mbase + j;
        float sg = fsigmoid(acc[j] + bias);
        if (gate == 0) zbuf[b * H_ + hc] = sg;
        else           rh[b * H_ + hc] = (__bf16)(sg * hf1[b * H_ + hc]);
      }
    }
    grid_sync(arrive, gen);

    // ---- Phase D: layer1 g + state update  (A=[h0_new | r*h1]) -----------
    for (int job = wid; job < 2 * 64; job += nw) {
      int mt = job & 1, nt3 = 128 + (job >> 1);
      v8f acc = {};
      #pragma unroll 4
      for (int f = 0; f < 32; ++f)
        acc = wmma_bf16(load_afrag(hb0, H_, mt * 16, f * 32),
                        load_pack(wx1p, nt3 * 32 + f), acc);
      #pragma unroll 4
      for (int kk = 0; kk < 32; ++kk)
        acc = wmma_bf16(load_afrag(rh, H_, mt * 16, kk * 32),
                        load_pack(wh1, nt3 * 32 + kk), acc);
      int hc = (nt3 - 128) * 16 + nloc;
      float bias = bh[5 * H_ + hc];
      #pragma unroll
      for (int j = 0; j < 8; ++j) {
        int b = mt * 16 + mbase + j;
        float g  = ftanh(acc[j] + bias);
        float z  = zbuf[b * H_ + hc];
        float hn = z * hf1[b * H_ + hc] + (1.0f - z) * g;
        hf1[b * H_ + hc] = hn;
        hb1[b * H_ + hc] = (__bf16)hn;
      }
    }
    grid_sync(arrive, gen);

    // ---- Phase E: output projection  y = h1_new @ Why^T + by -------------
    // (no barrier needed before next Phase A: disjoint buffers)
    for (int job = wid; job < 2 * 16; job += nw) {
      int mt = job & 1, nt = job >> 1;
      v8f acc = {};
      #pragma unroll 4
      for (int f = 0; f < 32; ++f)
        acc = wmma_bf16(load_afrag(hb1, H_, mt * 16, f * 32),
                        load_pack(whyp, nt * 32 + f), acc);
      int o = nt * 16 + nloc;
      float bias = by[o];
      #pragma unroll
      for (int j = 0; j < 8; ++j) {
        int b = mt * 16 + mbase + j;
        out[(size_t)b * (S_ * O_) + (size_t)s * O_ + o] = acc[j] + bias;
      }
    }
  }

  // ---- final hidden state [B,L,H] appended after [B,S,O] ------------------
  int stride = NBLOCKS * NTHREADS;
  for (int i = (int)(blockIdx.x * blockDim.x + threadIdx.x); i < B_ * L_ * H_;
       i += stride) {
    int b   = i >> 11;           // L*H = 2048
    int rem = i & 2047;
    int ly  = rem >> 10;
    int h   = rem & (H_ - 1);
    out[(size_t)B_ * S_ * O_ + i] = hf[(size_t)ly * (B_ * H_) + (size_t)b * H_ + h];
  }
}

// ------------------------------- launcher ----------------------------------
static inline int grids_for(long total) {
  long g = (total + NTHREADS - 1) / NTHREADS;
  if (g > 4096) g = 4096;
  return (int)g;
}

extern "C" void kernel_launch(void* const* d_in, const int* in_sizes, int n_in,
                              void* d_out, int out_size, void* d_ws, size_t ws_size,
                              hipStream_t stream) {
  (void)in_sizes; (void)n_in; (void)out_size; (void)ws_size;
  const float* x   = (const float*)d_in[0];
  const float* h0  = (const float*)d_in[1];
  const float* Wx0 = (const float*)d_in[2];   // [3,H,I]
  const float* Wx1 = (const float*)d_in[3];   // [1,3,H,H]
  const float* Wh  = (const float*)d_in[4];   // [2,3,H,H]
  const float* bh  = (const float*)d_in[5];   // [2,3,H]
  const float* Why = (const float*)d_in[6];   // [O,H]
  const float* by  = (const float*)d_in[7];   // [O]
  float* out = (float*)d_out;
  char*  ws  = (char*)d_ws;

  unsigned* bar  = (unsigned*)(ws + OFF_BAR);
  __bf16* wx0p = (__bf16*)(ws + OFF_WX0);
  __bf16* wx1p = (__bf16*)(ws + OFF_WX1);
  __bf16* whp  = (__bf16*)(ws + OFF_WH);
  __bf16* whyp = (__bf16*)(ws + OFF_WHY);
  __bf16* xp   = (__bf16*)(ws + OFF_XP);
  float*  hf   = (float*)(ws + OFF_HF);
  __bf16* hb   = (__bf16*)(ws + OFF_HB);
  float*  zbuf = (float*)(ws + OFF_Z);
  __bf16* rhb  = (__bf16*)(ws + OFF_RH);

  init_kernel<<<64, NTHREADS, 0, stream>>>(h0, hf, hb, bar);

  long t;
  t = (long)3 * H_ * I_;
  pack_b_kernel<<<grids_for(t), NTHREADS, 0, stream>>>(Wx0, wx0p, I_, 8, t);
  t = (long)3 * H_ * H_;
  pack_b_kernel<<<grids_for(t), NTHREADS, 0, stream>>>(Wx1, wx1p, H_, 32, t);
  t = (long)6 * H_ * H_;
  pack_b_kernel<<<grids_for(t), NTHREADS, 0, stream>>>(Wh, whp, H_, 32, t);
  t = (long)O_ * H_;
  pack_b_kernel<<<grids_for(t), NTHREADS, 0, stream>>>(Why, whyp, H_, 32, t);
  t = (long)B_ * S_ * I_;
  pack_x_kernel<<<grids_for(t), NTHREADS, 0, stream>>>(x, xp, t);

  gru_persistent<<<NBLOCKS, NTHREADS, 0, stream>>>(
      wx0p, wx1p, whp, whyp, xp, hf, hb, zbuf, rhb, bar, bh, by, out);
}
